// ExtraMSABlock_84396107366419
// MI455X (gfx1250) — matrix operand order repaired and checked by reading, not compiled, not run on hardware
//
#include <hip/hip_runtime.h>

#define DEV __device__ __forceinline__

typedef __attribute__((ext_vector_type(16))) _Float16 v16h;
typedef __attribute__((ext_vector_type(8)))  _Float16 v8h;
typedef __attribute__((ext_vector_type(8)))  float    v8f;

static constexpr int S_  = 128;
static constexpr int N_  = 256;
static constexpr int CE_ = 64;
static constexpr int CZ_ = 128;
static constexpr int SN_ = S_ * N_;   // 32768
static constexpr int NN_ = N_ * N_;   // 65536

DEV float fast_rcp(float x) { return __builtin_amdgcn_rcpf(x); }
DEV float fast_sigmoid(float x) { return fast_rcp(1.f + __expf(-x)); }

// ---------------- WMMA fragment index maps (per CDNA5 ISA tables) ----------
// A-matrix 16x32 f16: lanes0-15 hold K={0..7,16..23}, lanes16-31 K={8..15,24..31}
DEV int kmapA(int lane, int e) { return (e + (e & 8)) + ((lane >> 4) << 3); }
// B-matrix 32x16 f16: lanes0-15 hold K={0..7,16..23}, lanes16-31 K={8..15,24..31}
DEV int kmapB(int lane, int e) { return (e + (e & 8)) + ((lane >> 4) << 3); }

DEV v8f wmma_f16(v16h a, v16h b, v8f c) {
  return __builtin_amdgcn_wmma_f32_16x16x32_f16(false, a, false, b, (short)0, c,
                                                false, false);
}

// A-fragment from f32 row pointer already offset by (k0 + kq): two contiguous
// 8-float runs at +0 and +16 -> 4x b128 loads, no branches.
DEV v16h fragA_f32(const float* p) {
  float4 x0 = *(const float4*)(p);
  float4 x1 = *(const float4*)(p + 4);
  float4 y0 = *(const float4*)(p + 16);
  float4 y1 = *(const float4*)(p + 20);
  v16h f;
  f[0] = (_Float16)x0.x; f[1] = (_Float16)x0.y; f[2] = (_Float16)x0.z; f[3] = (_Float16)x0.w;
  f[4] = (_Float16)x1.x; f[5] = (_Float16)x1.y; f[6] = (_Float16)x1.z; f[7] = (_Float16)x1.w;
  f[8] = (_Float16)y0.x; f[9] = (_Float16)y0.y; f[10] = (_Float16)y0.z; f[11] = (_Float16)y0.w;
  f[12] = (_Float16)y1.x; f[13] = (_Float16)y1.y; f[14] = (_Float16)y1.z; f[15] = (_Float16)y1.w;
  return f;
}
// A-fragment from f16 row pointer already offset by (k0 + kq): 2x b128 loads.
DEV v16h fragA_h16(const _Float16* p) {
  v8h lo = *(const v8h*)(p);
  v8h hi = *(const v8h*)(p + 16);
  v16h f;
#pragma unroll
  for (int e = 0; e < 8; ++e) { f[e] = lo[e]; f[e + 8] = hi[e]; }
  return f;
}
// B-fragment from transposed f16 weights Bt[N][K]: two 8-half runs at +0 and
// +16 from (ncol*K + kbase) -> 2x b128 loads, zero converts in the hot loop.
DEV v16h fragB_h16(const _Float16* p) {
  v8h lo = *(const v8h*)(p);
  v8h hi = *(const v8h*)(p + 16);
  v16h f;
#pragma unroll
  for (int e = 0; e < 8; ++e) { f[e] = lo[e]; f[e + 8] = hi[e]; }
  return f;
}

// ---------------- weight prep: B[K,N] f32 -> Bt[N,K] f16 ---------------------
__global__ void k_prep_bt(const float* __restrict__ B, _Float16* __restrict__ Bt,
                          int K, int Nn) {
  int idx = blockIdx.x * blockDim.x + threadIdx.x;
  if (idx >= K * Nn) return;
  int n = idx % Nn, k = idx / Nn;
  Bt[(size_t)n * K + k] = (_Float16)B[idx];
}

// ---------------- LayerNorm: one wave per row --------------------------------
__global__ void k_ln(const float* __restrict__ x, const float* __restrict__ w,
                     const float* __restrict__ b, float* __restrict__ y,
                     int rows, int C) {
  int row = blockIdx.x * (blockDim.x >> 5) + (threadIdx.x >> 5);
  int lane = threadIdx.x & 31;
  if (row >= rows) return;
  const float* xr = x + (size_t)row * C;
  float s = 0.f, s2 = 0.f;
  for (int c = lane; c < C; c += 32) { float v = xr[c]; s += v; s2 += v * v; }
  for (int off = 16; off > 0; off >>= 1) {
    s  += __shfl_xor(s,  off, 32);
    s2 += __shfl_xor(s2, off, 32);
  }
  float mu  = s / (float)C;
  float var = s2 / (float)C - mu * mu;
  float inv = rsqrtf(var + 1e-5f);
  float* yr = y + (size_t)row * C;
  for (int c = lane; c < C; c += 32) yr[c] = (xr[c] - mu) * inv * w[c] + b[c];
}

// ---------------- WMMA GEMM: out = act(A[M,K]@B[K,N]*scale + bias) -----------
// B supplied pre-transposed as f16 Bt[N][K]. Requires M%32==0, N%16==0,
// K%32==0. One wave -> 32x16 output (2 m-tiles sharing each B fragment).
__global__ __launch_bounds__(256) void k_gemm(
    const float* __restrict__ A, const _Float16* __restrict__ Bt,
    const float* __restrict__ bias, float* __restrict__ C,
    int M, int K, int Nn, int act, float scale) {
  int lane = threadIdx.x & 31;
  int wave = blockIdx.x * (blockDim.x >> 5) + (threadIdx.x >> 5);
  int tn = Nn >> 4, tm = M >> 5;
  if (wave >= tm * tn) return;
  int mbase = (wave / tn) << 5, nbase = (wave % tn) << 4;
  int ncol = nbase + (lane & 15);
  int kq = (lane >> 4) << 3;          // A k-offset for this lane
  int kb = (lane >> 4) << 3;          // B k-offset for this lane
  const float* Ar0 = A + (size_t)(mbase + (lane & 15)) * K + kq;
  const float* Ar1 = Ar0 + (size_t)16 * K;
  const _Float16* Bp = Bt + (size_t)ncol * K + kb;
  v8f acc0 = {}, acc1 = {};
  for (int k0 = 0; k0 < K; k0 += 32) {
    v16h b  = fragB_h16(Bp + k0);
    v16h a0 = fragA_f32(Ar0 + k0);
    v16h a1 = fragA_f32(Ar1 + k0);
    acc0 = wmma_f16(a0, b, acc0);
    acc1 = wmma_f16(a1, b, acc1);
  }
  float bv = bias ? bias[ncol] : 0.f;
#pragma unroll
  for (int r = 0; r < 8; ++r) {
    int m = mbase + r + ((lane >> 4) << 3);
    float v0 = acc0[r] * scale + bv;
    float v1 = acc1[r] * scale + bv;
    if (act == 1) { v0 = fmaxf(v0, 0.f); v1 = fmaxf(v1, 0.f); }
    else if (act == 2) { v0 = fast_sigmoid(v0); v1 = fast_sigmoid(v1); }
    C[(size_t)m * Nn + ncol] = v0;
    C[(size_t)(m + 16) * Nn + ncol] = v1;
  }
}

// ---------------- small projection CE->8 (ca_wk / ca_wv) ---------------------
__global__ void k_proj8(const float* __restrict__ x, const float* __restrict__ w,
                        float* __restrict__ y) {
  int t = blockIdx.x * blockDim.x + threadIdx.x;
  if (t >= SN_ * 8) return;
  int row = t >> 3, c = t & 7;
  const float* xr = x + (size_t)row * 64;
  float s = 0.f;
  for (int k = 0; k < 64; ++k) s += xr[k] * w[k * 8 + c];
  y[t] = s;
}

// ---------------- head bias: out[h,i,j] = dot(zln[i,j,:], w[:,h]) ------------
__global__ void k_headbias(const float* __restrict__ zln, const float* __restrict__ w,
                           float* __restrict__ out, int C, int H) {
  int ij = blockIdx.x * blockDim.x + threadIdx.x;
  if (ij >= NN_) return;
  float acc[8];
  for (int h = 0; h < 8; ++h) acc[h] = 0.f;
  const float* zr = zln + (size_t)ij * C;
  for (int c = 0; c < C; ++c) {
    float x = zr[c];
    for (int h = 0; h < H; ++h) acc[h] += x * w[c * H + h];
  }
  for (int h = 0; h < H; ++h) out[(size_t)h * NN_ + ij] = acc[h];
}

// ---------------- MSA row attention (online softmax, c=8 per head) -----------
__global__ void k_msa_row_attn(const float* __restrict__ q, const float* __restrict__ kk,
                               const float* __restrict__ vv, const float* __restrict__ pb,
                               const float* __restrict__ mask, float* __restrict__ o) {
  int t = blockIdx.x * blockDim.x + threadIdx.x;
  if (t >= S_ * 8 * N_) return;
  int i = t % N_, h = (t / N_) % 8, s = t / (N_ * 8);
  const float* qp  = q  + ((size_t)s * N_ + i) * 64 + h * 8;
  const float* pbr = pb + ((size_t)h * N_ + i) * N_;
  const float* mr  = mask + (size_t)s * N_;
  float mx = -3.0e38f, ssum = 0.f, oa[8];
  for (int c = 0; c < 8; ++c) oa[c] = 0.f;
  for (int j = 0; j < N_; ++j) {
    const float* kp = kk + ((size_t)s * N_ + j) * 64 + h * 8;
    float l = 0.f;
    for (int c = 0; c < 8; ++c) l += qp[c] * kp[c];
    l += 1e9f * (mr[j] - 1.f) + pbr[j];
    float mn = fmaxf(mx, l);
    float corr = __expf(mx - mn), wgt = __expf(l - mn);
    ssum = ssum * corr + wgt;
    const float* vp = vv + ((size_t)s * N_ + j) * 64 + h * 8;
    for (int c = 0; c < 8; ++c) oa[c] = oa[c] * corr + wgt * vp[c];
    mx = mn;
  }
  float inv = fast_rcp(ssum);
  float* op = o + ((size_t)s * N_ + i) * 64 + h * 8;
  for (int c = 0; c < 8; ++c) op[c] = oa[c] * inv;
}

// ---------------- MSA column-global attention helpers ------------------------
__global__ void k_qavg(const float* __restrict__ mln, const float* __restrict__ mask,
                       float* __restrict__ out) {
  int t = blockIdx.x * blockDim.x + threadIdx.x;
  if (t >= N_ * 64) return;
  int i = t >> 6, c = t & 63;
  float num = 0.f, den = 0.f;
  for (int s = 0; s < S_; ++s) {
    float mk = mask[(size_t)s * N_ + i];
    num += mln[((size_t)s * N_ + i) * 64 + c] * mk;
    den += mk;
  }
  out[t] = num / (den + 1e-8f);
}

__global__ void k_ca_attn(const float* __restrict__ q, const float* __restrict__ kk,
                          const float* __restrict__ vv, const float* __restrict__ mask,
                          float* __restrict__ o) {
  int t = blockIdx.x * blockDim.x + threadIdx.x;
  if (t >= N_ * 8) return;
  int i = t >> 3, h = t & 7;
  const float* qp = q + (size_t)i * 64 + h * 8;
  float mx = -3.0e38f, ssum = 0.f, oa[8];
  for (int c = 0; c < 8; ++c) oa[c] = 0.f;
  for (int s = 0; s < S_; ++s) {
    const float* kp = kk + ((size_t)s * N_ + i) * 8;
    float l = 0.f;
    for (int c = 0; c < 8; ++c) l += qp[c] * kp[c];
    l += 1e9f * (mask[(size_t)s * N_ + i] - 1.f);
    float mn = fmaxf(mx, l);
    float corr = __expf(mx - mn), wgt = __expf(l - mn);
    ssum = ssum * corr + wgt;
    const float* vp = vv + ((size_t)s * N_ + i) * 8;
    for (int c = 0; c < 8; ++c) oa[c] = oa[c] * corr + wgt * vp[c];
    mx = mn;
  }
  float inv = fast_rcp(ssum);
  for (int c = 0; c < 8; ++c) o[(size_t)i * 64 + h * 8 + c] = oa[c] * inv;
}

// ---------------- elementwise helpers ---------------------------------------
__global__ void k_mul(float* __restrict__ a, const float* __restrict__ b, int n) {
  int i = blockIdx.x * blockDim.x + threadIdx.x;
  if (i < n) a[i] *= b[i];
}
__global__ void k_mul_bcast(float* __restrict__ out, const float* __restrict__ g,
                            const float* __restrict__ o, int n, int period) {
  int i = blockIdx.x * blockDim.x + threadIdx.x;
  if (i < n) out[i] = g[i] * o[i % period];
}
__global__ void k_mask_mul(float* __restrict__ a, const float* __restrict__ mask,
                           int C, int n) {
  int i = blockIdx.x * blockDim.x + threadIdx.x;
  if (i < n) a[i] *= mask[i / C];
}
__global__ void k_gate_mask(float* __restrict__ p, const float* __restrict__ g,
                            const float* __restrict__ mask, int C, int n) {
  int i = blockIdx.x * blockDim.x + threadIdx.x;
  if (i < n) p[i] = p[i] * g[i] * mask[i / C];
}
__global__ void k_add(float* __restrict__ dst, const float* __restrict__ src, int n) {
  int i = blockIdx.x * blockDim.x + threadIdx.x;
  if (i < n) dst[i] += src[i];
}
__global__ void k_addT(float* __restrict__ Z, const float* __restrict__ src) {
  int idx = blockIdx.x * blockDim.x + threadIdx.x;
  if (idx >= NN_ * CZ_) return;
  int c = idx & 127, ij = idx >> 7;
  int j = ij & 255, i = ij >> 8;
  Z[idx] += src[(((size_t)j * N_ + i) << 7) + c];
}
__global__ void k_transpose(const float* __restrict__ src, float* __restrict__ dst, int C) {
  long long idx = (long long)blockIdx.x * blockDim.x + threadIdx.x;
  long long total = (long long)NN_ * C;
  if (idx >= total) return;
  int c = (int)(idx % C);
  int ij = (int)(idx / C);
  int j = ij & 255, i = ij >> 8;
  dst[((size_t)j * N_ + i) * C + c] = src[idx];
}
__global__ void k_norm(const float* __restrict__ mask, float* __restrict__ out) {
  int ij = blockIdx.x * blockDim.x + threadIdx.x;
  if (ij >= NN_) return;
  int j = ij & 255, i = ij >> 8;
  float s = 0.f;
  for (int ss = 0; ss < S_; ++ss)
    s += mask[(size_t)ss * N_ + i] * mask[(size_t)ss * N_ + j];
  out[ij] = s;
}

// ---------------- Fused outer-product-mean -----------------------------------
// WG(256 thr) per (i, 16-j chunk). Phase1: P[c=32][(jl,d)=512] = sum_s a*b (WMMA,
// staged f16 in LDS). Phase2: out[16,128] = P(16x1024) @ wo (WMMA from LDS,
// wo pre-transposed to f16 [128][1024]).
__global__ __launch_bounds__(256) void k_opm_fused(
    const float* __restrict__ a, const float* __restrict__ b,
    const _Float16* __restrict__ wot, const float* __restrict__ bo,
    const float* __restrict__ nrm, float* __restrict__ Z) {
  __shared__ _Float16 P[16 * 1024];
  int i = blockIdx.y, j0 = blockIdx.x * 16;
  int lane = threadIdx.x & 31, wave = threadIdx.x >> 5;
  int mt = wave & 1, ntb = (wave >> 1) * 8;
  v8f zero = {};
  v8f acc[8];
#pragma unroll
  for (int t = 0; t < 8; ++t) acc[t] = zero;
  for (int k0 = 0; k0 < 128; k0 += 32) {
    v16h af;
    {
      int m = mt * 16 + (lane & 15);
#pragma unroll
      for (int e = 0; e < 16; ++e) {
        int s = k0 + kmapA(lane, e);
        af[e] = (_Float16)a[((size_t)s * N_ + i) * 32 + m];
      }
    }
#pragma unroll
    for (int t = 0; t < 8; ++t) {
      int n = (ntb + t) * 16 + (lane & 15);
      int jl = n >> 5, d = n & 31;
      v16h bf;
#pragma unroll
      for (int e = 0; e < 16; ++e) {
        int s = k0 + kmapB(lane, e);
        bf[e] = (_Float16)b[((size_t)s * N_ + (j0 + jl)) * 32 + d];
      }
      acc[t] = wmma_f16(af, bf, acc[t]);
    }
  }
#pragma unroll
  for (int t = 0; t < 8; ++t) {
    int n = (ntb + t) * 16 + (lane & 15);
    int jl = n >> 5, d = n & 31;
#pragma unroll
    for (int r = 0; r < 8; ++r) {
      int m = mt * 16 + r + ((lane >> 4) << 3);
      P[jl * 1024 + m * 32 + d] = (_Float16)acc[t][r];
    }
  }
  __syncthreads();
  v8f acc2 = zero;
  int nb = wave * 16;
  int ml = lane & 15;
  const _Float16* Prow = P + ml * 1024 + ((lane >> 4) << 3);
  const _Float16* Wp = wot + (size_t)(nb + ml) * 1024 + ((lane >> 4) << 3);
  for (int k0 = 0; k0 < 1024; k0 += 32) {
    v16h af = fragA_h16(Prow + k0);
    v16h bf = fragB_h16(Wp + k0);
    acc2 = wmma_f16(af, bf, acc2);
  }
  int n = nb + ml;
#pragma unroll
  for (int r = 0; r < 8; ++r) {
    int jl = r + ((lane >> 4) << 3);
    int j = j0 + jl;
    float val = (acc2[r] + bo[n]) * fast_rcp(nrm[(size_t)i * N_ + j] + 1e-3f);
    Z[((size_t)i * N_ + j) * 128 + n] += val;
  }
}

// ---------------- Triangle multiply einsum: per-channel batched GEMM ---------
__global__ __launch_bounds__(256) void k_tri_einsum(
    const float* __restrict__ a, const float* __restrict__ b,
    float* __restrict__ x, int incoming) {
  int lane = threadIdx.x & 31;
  int wave = blockIdx.x * (blockDim.x >> 5) + (threadIdx.x >> 5);
  if (wave >= 128 * 256) return;
  int c = wave >> 8;
  int rem = wave & 255;
  int mbase = (rem >> 4) << 4, nbase = (rem & 15) << 4;
  int mi = mbase + (lane & 15), ni = nbase + (lane & 15);
  v8f acc = {};
  for (int k0 = 0; k0 < N_; k0 += 32) {
    v16h af, bf;
#pragma unroll
    for (int e = 0; e < 16; ++e) {
      int ka = k0 + kmapA(lane, e);
      int kb = k0 + kmapB(lane, e);
      float av = incoming ? a[((size_t)ka * N_ + mi) * 128 + c]
                          : a[((size_t)mi * N_ + ka) * 128 + c];
      float bv = incoming ? b[((size_t)kb * N_ + ni) * 128 + c]
                          : b[((size_t)ni * N_ + kb) * 128 + c];
      af[e] = (_Float16)av;
      bf[e] = (_Float16)bv;
    }
    acc = wmma_f16(af, bf, acc);
  }
#pragma unroll
  for (int r = 0; r < 8; ++r) {
    int m = mbase + r + ((lane >> 4) << 3);
    x[((size_t)m * N_ + nbase + (lane & 15)) * 128 + c] = acc[r];
  }
}

// ---------------- Triangle attention -----------------------------------------
// WG(128 thr) per (i, h, 16-row j tile): WMMA logits(K=32) -> LDS f32 ->
// softmax -> WMMA A@V(K=256) from LDS.
__global__ __launch_bounds__(128) void k_tri_attn(
    const float* __restrict__ q, const float* __restrict__ kk,
    const float* __restrict__ vv, const float* __restrict__ tb,
    const float* __restrict__ mask, float* __restrict__ o) {
  __shared__ float L[16 * 256];
  int jt = blockIdx.x, h = blockIdx.y, i = blockIdx.z;
  int lane = threadIdx.x & 31, wave = threadIdx.x >> 5;
  int jb = jt * 16;
  v16h af = fragA_f32(q + ((size_t)i * N_ + jb + (lane & 15)) * 128 + h * 32 +
                      ((lane >> 4) << 3));
  for (int t = 0; t < 4; ++t) {
    int nbase = (wave * 4 + t) * 16;
    int n = nbase + (lane & 15);
    v16h bf;
#pragma unroll
    for (int e = 0; e < 16; ++e)
      bf[e] = (_Float16)kk[((size_t)i * N_ + n) * 128 + h * 32 + kmapB(lane, e)];
    v8f acc = {};
    acc = wmma_f16(af, bf, acc);
#pragma unroll
    for (int r = 0; r < 8; ++r) {
      int m = r + ((lane >> 4) << 3);
      L[m * 256 + n] = acc[r] + 1e9f * (mask[(size_t)i * N_ + n] - 1.f)
                     + tb[((size_t)h * N_ + jb + m) * N_ + n];
    }
  }
  __syncthreads();
  if (threadIdx.x < 16) {
    float* row = L + threadIdx.x * 256;
    float mx = -3.0e38f;
    for (int j = 0; j < 256; ++j) mx = fmaxf(mx, row[j]);
    float s = 0.f;
    for (int j = 0; j < 256; ++j) { float e = __expf(row[j] - mx); row[j] = e; s += e; }
    float inv = fast_rcp(s);
    for (int j = 0; j < 256; ++j) row[j] *= inv;
  }
  __syncthreads();
  if (wave < 2) {
    v8f acc = {};
    int nb = wave * 16, ml = lane & 15;
    const float* Lrow = L + ml * 256 + ((lane >> 4) << 3);
    for (int k0 = 0; k0 < 256; k0 += 32) {
      v16h a2 = fragA_f32(Lrow + k0);
      v16h b2;
#pragma unroll
      for (int e = 0; e < 16; ++e)
        b2[e] = (_Float16)vv[((size_t)i * N_ + k0 + kmapB(lane, e)) * 128 + h * 32 + nb + ml];
      acc = wmma_f16(a2, b2, acc);
    }
#pragma unroll
    for (int r = 0; r < 8; ++r) {
      int m = r + ((lane >> 4) << 3);
      o[((size_t)i * N_ + jb + m) * 128 + h * 32 + nb + ml] = acc[r];
    }
  }
}

// =============================== host side ===================================
extern "C" void kernel_launch(void* const* d_in, const int* in_sizes, int n_in,
                              void* d_out, int out_size, void* d_ws, size_t ws_size,
                              hipStream_t stream) {
  (void)in_sizes; (void)n_in; (void)out_size; (void)ws_size;
  const float* in_m      = (const float*)d_in[0];
  const float* in_z      = (const float*)d_in[1];
  const float* msa_mask  = (const float*)d_in[2];
  const float* pair_mask = (const float*)d_in[3];
#define PRM(i) ((const float*)d_in[4 + (i)])

  float* M_ = (float*)d_out;
  float* Z_ = M_ + (size_t)SN_ * CE_;
  float* ws = (float*)d_ws;
  float* t0 = ws;                 // 32 MB arenas
  float* t1 = ws + 8388608;
  float* t2 = ws + 16777216;
  float* t3 = ws + 25165824;
  float* t4 = ws + 33554432;
  float* t5 = ws + 41943040;
  float* t6 = ws + 50331648;      // 8 MB: mln
  float* t7 = ws + 52428800;      // 4 MB: pb/tb + misc
  float* qavg  = t7 + 524288;
  float* qca   = t7 + 540672;
  float* oca   = t7 + 557056;
  float* nrm   = t7 + 573440;
  float* maskT = t7 + 638976;
  _Float16* bt = (_Float16*)(ws + 53477376);  // 256 KB f16 weight scratch

  hipMemcpyAsync(M_, in_m, (size_t)SN_ * CE_ * sizeof(float),
                 hipMemcpyDeviceToDevice, stream);
  hipMemcpyAsync(Z_, in_z, (size_t)NN_ * CZ_ * sizeof(float),
                 hipMemcpyDeviceToDevice, stream);

  auto grid1 = [](int n) { return dim3((n + 255) / 256); };
  auto ln = [&](const float* x, const float* w, const float* b, float* y,
                int rows, int C) {
    k_ln<<<dim3((rows + 7) / 8), dim3(256), 0, stream>>>(x, w, b, y, rows, C);
  };
  auto gemm = [&](const float* A, const float* B, const float* bias, float* C,
                  int M, int K, int Nn, int act = 0, float scale = 1.f) {
    k_prep_bt<<<grid1(K * Nn), 256, 0, stream>>>(B, bt, K, Nn);
    int tiles = (M >> 5) * (Nn >> 4);
    k_gemm<<<dim3((tiles + 7) / 8), dim3(256), 0, stream>>>(A, bt, bias, C, M, K,
                                                            Nn, act, scale);
  };

  // ---- 1. MSA row attention with pair bias --------------------------------
  ln(M_, PRM(0), PRM(1), t6, SN_, CE_);
  ln(Z_, PRM(2), PRM(3), t0, NN_, CZ_);
  k_headbias<<<grid1(NN_), 256, 0, stream>>>(t0, PRM(4), t7, CZ_, 8);
  gemm(t6, PRM(5), nullptr, t1, SN_, CE_, 64, 0, 0.35355339f); // q
  gemm(t6, PRM(6), nullptr, t2, SN_, CE_, 64);                 // k
  gemm(t6, PRM(7), nullptr, t3, SN_, CE_, 64);                 // v
  gemm(t6, PRM(8), PRM(9), t4, SN_, CE_, 64, 2);               // g
  k_msa_row_attn<<<grid1(S_ * 8 * N_), 256, 0, stream>>>(t1, t2, t3, t7,
                                                         msa_mask, t5);
  k_mul<<<grid1(SN_ * 64), 256, 0, stream>>>(t5, t4, SN_ * 64);
  gemm(t5, PRM(10), PRM(11), t1, SN_, 64, CE_);
  k_add<<<grid1(SN_ * CE_), 256, 0, stream>>>(M_, t1, SN_ * CE_);

  // ---- 2. MSA column global attention -------------------------------------
  ln(M_, PRM(12), PRM(13), t6, SN_, CE_);
  k_qavg<<<grid1(N_ * 64), 256, 0, stream>>>(t6, msa_mask, qavg);
  gemm(qavg, PRM(14), nullptr, qca, N_, 64, 64, 0, 0.35355339f);
  k_proj8<<<grid1(SN_ * 8), 256, 0, stream>>>(t6, PRM(15), t1);  // k
  k_proj8<<<grid1(SN_ * 8), 256, 0, stream>>>(t6, PRM(16), t2);  // v
  gemm(t6, PRM(17), PRM(18), t3, SN_, 64, 64, 2);                // g
  k_ca_attn<<<grid1(N_ * 8), 256, 0, stream>>>(qca, t1, t2, msa_mask, oca);
  k_mul_bcast<<<grid1(SN_ * 64), 256, 0, stream>>>(t4, t3, oca, SN_ * 64,
                                                   N_ * 64);
  gemm(t4, PRM(19), PRM(20), t5, SN_, 64, CE_);
  k_add<<<grid1(SN_ * CE_), 256, 0, stream>>>(M_, t5, SN_ * CE_);

  // ---- 3. MSA transition ---------------------------------------------------
  ln(M_, PRM(21), PRM(22), t6, SN_, CE_);
  gemm(t6, PRM(23), PRM(24), t1, SN_, 64, 256, 1);
  gemm(t1, PRM(25), PRM(26), t2, SN_, 256, 64);
  k_add<<<grid1(SN_ * CE_), 256, 0, stream>>>(M_, t2, SN_ * CE_);

  // ---- 4. Outer product mean ----------------------------------------------
  ln(M_, PRM(27), PRM(28), t6, SN_, CE_);
  gemm(t6, PRM(29), PRM(30), t1, SN_, 64, 32);
  k_mask_mul<<<grid1(SN_ * 32), 256, 0, stream>>>(t1, msa_mask, 32, SN_ * 32);
  gemm(t6, PRM(31), PRM(32), t2, SN_, 64, 32);
  k_mask_mul<<<grid1(SN_ * 32), 256, 0, stream>>>(t2, msa_mask, 32, SN_ * 32);
  k_norm<<<grid1(NN_), 256, 0, stream>>>(msa_mask, nrm);
  k_prep_bt<<<grid1(1024 * 128), 256, 0, stream>>>(PRM(33), bt, 1024, 128);
  k_opm_fused<<<dim3(16, 256), 256, 0, stream>>>(t1, t2, bt, PRM(34), nrm, Z_);

  // ---- 5/6. Triangle multiplies -------------------------------------------
  auto trimul = [&](int base, int incoming) {
    ln(Z_, PRM(base + 0), PRM(base + 1), t0, NN_, CZ_);
    gemm(t0, PRM(base + 2), PRM(base + 3), t1, NN_, CZ_, 128);     // a proj
    gemm(t0, PRM(base + 4), PRM(base + 5), t2, NN_, CZ_, 128, 2);  // a gate
    k_gate_mask<<<grid1(NN_ * 128), 256, 0, stream>>>(t1, t2, pair_mask, 128,
                                                      NN_ * 128);
    gemm(t0, PRM(base + 6), PRM(base + 7), t2, NN_, CZ_, 128);     // b proj
    gemm(t0, PRM(base + 8), PRM(base + 9), t3, NN_, CZ_, 128, 2);  // b gate
    k_gate_mask<<<grid1(NN_ * 128), 256, 0, stream>>>(t2, t3, pair_mask, 128,
                                                      NN_ * 128);
    k_tri_einsum<<<dim3(4096), 256, 0, stream>>>(t1, t2, t3, incoming);
    ln(t3, PRM(base + 10), PRM(base + 11), t4, NN_, 128);
    gemm(t0, PRM(base + 12), PRM(base + 13), t5, NN_, CZ_, CZ_, 2); // g
    gemm(t4, PRM(base + 14), PRM(base + 15), t1, NN_, 128, CZ_);
    k_mul<<<grid1(NN_ * CZ_), 256, 0, stream>>>(t1, t5, NN_ * CZ_);
    k_add<<<grid1(NN_ * CZ_), 256, 0, stream>>>(Z_, t1, NN_ * CZ_);
  };
  trimul(35, 0);  // outgoing
  trimul(51, 1);  // incoming

  // ---- 7/8. Triangle attention (start / end) -------------------------------
  auto triattn = [&](int base, const float* zsrc, const float* pmask,
                     bool transposed) {
    ln(zsrc, PRM(base + 0), PRM(base + 1), t0, NN_, CZ_);
    k_headbias<<<grid1(NN_), 256, 0, stream>>>(t0, PRM(base + 5), t7, CZ_, 4);
    gemm(t0, PRM(base + 2), nullptr, t1, NN_, CZ_, 128, 0, 0.17677670f); // q
    gemm(t0, PRM(base + 3), nullptr, t2, NN_, CZ_, 128);                 // k
    gemm(t0, PRM(base + 4), nullptr, t3, NN_, CZ_, 128);                 // v
    gemm(t0, PRM(base + 6), PRM(base + 7), t4, NN_, CZ_, 128, 2);        // g
    k_tri_attn<<<dim3(16, 4, 256), 128, 0, stream>>>(t1, t2, t3, t7, pmask, t5);
    k_mul<<<grid1(NN_ * 128), 256, 0, stream>>>(t5, t4, NN_ * 128);
    gemm(t5, PRM(base + 8), PRM(base + 9), t1, NN_, 128, CZ_);
    if (!transposed)
      k_add<<<grid1(NN_ * CZ_), 256, 0, stream>>>(Z_, t1, NN_ * CZ_);
    else
      k_addT<<<grid1(NN_ * CZ_), 256, 0, stream>>>(Z_, t1);
  };
  triattn(67, Z_, pair_mask, false);          // starting node
  k_transpose<<<grid1(NN_ * 128), 256, 0, stream>>>(Z_, t5, 128);
  k_transpose<<<grid1(NN_), 256, 0, stream>>>(pair_mask, maskT, 1);
  triattn(77, t5, maskT, true);               // ending node (on z^T)

  // ---- 9. Pair transition --------------------------------------------------
  ln(Z_, PRM(87), PRM(88), t0, NN_, CZ_);
  gemm(t0, PRM(89), PRM(90), t1, NN_, 128, 512, 1);  // t1 spans 128MB (t1..t4)
  gemm(t1, PRM(91), PRM(92), t5, NN_, 512, 128);
  k_add<<<grid1(NN_ * CZ_), 256, 0, stream>>>(Z_, t5, NN_ * CZ_);
#undef PRM
}
